// Decoder_81063212745440
// MI455X (gfx1250) — compile-verified, matrix operand order checked by reading, not run on hardware
//
#include <hip/hip_runtime.h>
#include <hip/hip_bf16.h>
#include <cstdint>

// ---------------------------------------------------------------------------
// DCN pointer-decoder on gfx1250: bf16 WMMA (v_wmma_f32_16x16x32_bf16) for the
// expert einsum and HMN GEMM, maxout folded into per-wave register max.
// ---------------------------------------------------------------------------

typedef __attribute__((ext_vector_type(16))) __bf16 v16bf;
typedef __attribute__((ext_vector_type(8)))  float  v8f;

#define DIMD   128
#define MLEN   4096
#define NSTEPS 4

union FragBF {
    v16bf v;
    uint4 q[2];
};

__device__ __forceinline__ float sigmoidf_(float x) { return 1.0f / (1.0f + __expf(-x)); }

// --------------------------------------------------------------------------
// Kernel P: convert U, We_s, We_e, W2_s, W2_e to bf16 in workspace; reset
// h, c, si, ei each call (kernel_launch must be deterministic).
// --------------------------------------------------------------------------
__global__ void dec_precvt(const float* __restrict__ U,
                           const float* __restrict__ WeS, const float* __restrict__ WeE,
                           const float* __restrict__ W2S, const float* __restrict__ W2E,
                           __bf16* __restrict__ Ubf,
                           __bf16* __restrict__ WeSbf, __bf16* __restrict__ WeEbf,
                           __bf16* __restrict__ W2Sbf, __bf16* __restrict__ W2Ebf,
                           float* __restrict__ h, float* __restrict__ c,
                           int* __restrict__ idx) {
    const int nU  = MLEN * 2 * DIMD;      // 1048576
    const int nWe = 2 * 1024 * 384;       // 786432
    const int nW2 = 1024 * DIMD;          // 131072
    int gid = blockIdx.x * blockDim.x + threadIdx.x;
    int total = nU + 2 * nWe + 2 * nW2;
    for (int i = gid; i < total; i += gridDim.x * blockDim.x) {
        int j = i;
        if (j < nU)            { Ubf[j]   = (__bf16)U[j];   continue; }
        j -= nU;
        if (j < nWe)           { WeSbf[j] = (__bf16)WeS[j]; continue; }
        j -= nWe;
        if (j < nWe)           { WeEbf[j] = (__bf16)WeE[j]; continue; }
        j -= nWe;
        if (j < nW2)           { W2Sbf[j] = (__bf16)W2S[j]; continue; }
        j -= nW2;
        W2Ebf[j] = (__bf16)W2E[j];
    }
    if (gid < DIMD) { h[gid] = 0.0f; c[gid] = 0.0f; }
    if (gid == 0)   { idx[0] = 0; idx[1] = MLEN - 1; }
}

// --------------------------------------------------------------------------
// Kernel A (1 block, 256 thr): gather us/ue via si/ei, LSTM cell, r_s/r_e
// (tanh GEMV, stored bf16), softmax gates. All tiny scalar GEMVs.
// --------------------------------------------------------------------------
__global__ void dec_step_small(const float* __restrict__ U,
                               const float* __restrict__ Wih, const float* __restrict__ Whh,
                               const float* __restrict__ bih, const float* __restrict__ bhh,
                               const float* __restrict__ WrS, const float* __restrict__ brS,
                               const float* __restrict__ WgS, const float* __restrict__ bgS,
                               const float* __restrict__ WrE, const float* __restrict__ brE,
                               const float* __restrict__ WgE, const float* __restrict__ bgE,
                               float* __restrict__ h, float* __restrict__ c,
                               __bf16* __restrict__ rSbf, __bf16* __restrict__ rEbf,
                               float* __restrict__ gateS, float* __restrict__ gateE,
                               const int* __restrict__ idx) {
    __shared__ float xs[512];    // [us | ue]
    __shared__ float hs[DIMD];
    __shared__ float gs[512];
    __shared__ float ctx[640];   // [h | us | ue]
    __shared__ float glog[2][2];

    const int t  = threadIdx.x;
    const int si = idx[0];
    const int ei = idx[1];

    xs[t]       = U[si * 256 + t];
    xs[256 + t] = U[ei * 256 + t];
    if (t < DIMD) hs[t] = h[t];
    __syncthreads();

    // LSTM gate pre-activations: 512 rows, 2 per thread
    for (int j = t; j < 512; j += 256) {
        float acc = bih[j] + bhh[j];
        const float* wi = Wih + (size_t)j * 512;
        const float* wh = Whh + (size_t)j * DIMD;
        #pragma unroll 4
        for (int f = 0; f < 512; ++f) acc += wi[f] * xs[f];
        #pragma unroll 4
        for (int f = 0; f < DIMD; ++f) acc += wh[f] * hs[f];
        gs[j] = acc;
    }
    __syncthreads();

    if (t < DIMD) {
        float gi = gs[t], gf = gs[DIMD + t], gg = gs[2 * DIMD + t], go = gs[3 * DIMD + t];
        float c2 = sigmoidf_(gf) * c[t] + sigmoidf_(gi) * tanhf(gg);
        float h2 = sigmoidf_(go) * tanhf(c2);
        c[t] = c2; h[t] = h2; ctx[t] = h2;
    }
    ctx[DIMD + t]       = xs[t];         // us
    ctx[DIMD + 256 + t] = xs[256 + t];   // ue
    __syncthreads();

    // r = tanh(Wr @ ctx + br) for both sides (thread t: side=t>>7, row=t&127)
    {
        int side = t >> 7, j = t & 127;
        const float* Wr = side ? WrE : WrS;
        const float* br = side ? brE : brS;
        float acc = br[j];
        const float* wr = Wr + (size_t)j * 640;
        #pragma unroll 4
        for (int f = 0; f < 640; ++f) acc += wr[f] * ctx[f];
        __bf16 rb = (__bf16)tanhf(acc);
        if (side) rEbf[j] = rb; else rSbf[j] = rb;
    }
    // gate logits
    if (t < 4) {
        int side = t >> 1, kk = t & 1;
        const float* Wg = side ? WgE : WgS;
        const float* bg = side ? bgE : bgS;
        float acc = bg[kk];
        const float* wg = Wg + (size_t)kk * 640;
        for (int f = 0; f < 640; ++f) acc += wg[f] * ctx[f];
        glog[side][kk] = acc;
    }
    __syncthreads();
    if (t < 2) {
        float l0 = glog[t][0], l1 = glog[t][1];
        float mx = fmaxf(l0, l1);
        float e0 = __expf(l0 - mx), e1 = __expf(l1 - mx);
        float inv = 1.0f / (e0 + e1);
        float* g = t ? gateE : gateS;
        g[0] = e0 * inv; g[1] = e1 * inv;
    }
}

// inp[m, f] with f<256 -> Ubf, f>=256 -> broadcast r (chunks of 8 never straddle)
__device__ __forceinline__ uint4 inp_chunk(const __bf16* __restrict__ Ubf,
                                           const __bf16* __restrict__ rbf,
                                           int row, int f) {
    const __bf16* p = (f < 256) ? (Ubf + (size_t)row * 256 + f) : (rbf + (f - 256));
    return *(const uint4*)p;
}

// --------------------------------------------------------------------------
// Kernel B (grid=256 blocks of 16 rows, 256 thr = 8 waves):
//   phase 1: z = [U|r] @ We[k]^T + be  (K=384, 12 WMMA iters/tile),
//            wave w owns col tiles {w, w+8..w+56} == one maxout group ->
//            register max, then 2-expert gate mixture in registers -> m1.
//   phase 2: m2 = maxout(m1 @ W2^T + b2) with the same trick (K=128).
//   phase 3: scores[row] = max_q([m1|m2] @ W3^T + b3).
// --------------------------------------------------------------------------
__global__ void __launch_bounds__(256)
dec_expert_hmn(const __bf16* __restrict__ Ubf, const __bf16* __restrict__ rbf,
               const __bf16* __restrict__ Webf, const float* __restrict__ gate,
               const float* __restrict__ be,
               const __bf16* __restrict__ W2bf, const float* __restrict__ b2,
               const float* __restrict__ W3, const float* __restrict__ b3,
               float* __restrict__ outRow) {
    __shared__ __align__(16) float  m1f[16 * DIMD];
    __shared__ __align__(16) __bf16 m1bf[16 * DIMD];
    __shared__ __align__(16) float  m2f[16 * DIMD];
    __shared__ float sc[16 * 8];

    const int wave  = threadIdx.x >> 5;
    const int lane  = threadIdx.x & 31;
    const int ln    = lane & 15;     // column-within-tile / A row
    const int khalf = lane >> 4;     // which K half this lane holds
    const int rowbase = blockIdx.x * 16;
    const int arow    = rowbase + ln;

    const float g0 = gate[0], g1 = gate[1];

    // ---- phase 1: expert GEMMs + maxout + mixture --------------------------
    float vm[2][8];
    #pragma unroll
    for (int k = 0; k < 2; ++k) {
        #pragma unroll
        for (int j = 0; j < 8; ++j) vm[k][j] = -3.0e30f;
        const __bf16* Wek = Webf + (size_t)k * 1024 * 384;
        for (int pi = 0; pi < 8; ++pi) {
            const int ebase = (wave + 8 * pi) * 16;
            v8f acc = {};
            const __bf16* brow = Wek + (size_t)(ebase + ln) * 384;
            #pragma unroll
            for (int kb = 0; kb < 12; ++kb) {
                const int f0 = kb * 32 + khalf * 8;
                FragBF a, b;
                a.q[0] = inp_chunk(Ubf, rbf, arow, f0);
                a.q[1] = inp_chunk(Ubf, rbf, arow, f0 + 16);
                b.q[0] = *(const uint4*)(brow + f0);
                b.q[1] = *(const uint4*)(brow + f0 + 16);
                acc = __builtin_amdgcn_wmma_f32_16x16x32_bf16(
                        false, a.v, false, b.v, (short)0, acc, false, false);
            }
            const float bias = be[k * 1024 + ebase + ln];
            #pragma unroll
            for (int j = 0; j < 8; ++j) vm[k][j] = fmaxf(vm[k][j], acc[j] + bias);
        }
    }
    {
        const int jd = 16 * wave + ln;          // output column (e % 128)
        #pragma unroll
        for (int j = 0; j < 8; ++j) {
            const int rloc = khalf * 8 + j;     // C-layout row mapping
            float mval = g0 * vm[0][j] + g1 * vm[1][j];
            m1f[rloc * DIMD + jd]  = mval;
            m1bf[rloc * DIMD + jd] = (__bf16)mval;
        }
    }
    __syncthreads();

    // ---- phase 2: HMN GEMM m1 @ W2^T + b2, maxout over p=8 -----------------
    float v2[8];
    #pragma unroll
    for (int j = 0; j < 8; ++j) v2[j] = -3.0e30f;
    for (int pi = 0; pi < 8; ++pi) {
        const int ebase = (wave + 8 * pi) * 16;
        v8f acc = {};
        const __bf16* w2row = W2bf + (size_t)(ebase + ln) * DIMD;
        #pragma unroll
        for (int kb = 0; kb < 4; ++kb) {
            const int f0 = kb * 32 + khalf * 8;
            FragBF a, b;
            a.q[0] = *(const uint4*)(m1bf + ln * DIMD + f0);        // ds_load_b128
            a.q[1] = *(const uint4*)(m1bf + ln * DIMD + f0 + 16);
            b.q[0] = *(const uint4*)(w2row + f0);
            b.q[1] = *(const uint4*)(w2row + f0 + 16);
            acc = __builtin_amdgcn_wmma_f32_16x16x32_bf16(
                    false, a.v, false, b.v, (short)0, acc, false, false);
        }
        const float bias = b2[ebase + ln];
        #pragma unroll
        for (int j = 0; j < 8; ++j) v2[j] = fmaxf(v2[j], acc[j] + bias);
    }
    {
        const int jd = 16 * wave + ln;
        #pragma unroll
        for (int j = 0; j < 8; ++j) m2f[(khalf * 8 + j) * DIMD + jd] = v2[j];
    }
    __syncthreads();

    // ---- phase 3: score rows: max_q([m1|m2] @ W3^T + b3) -------------------
    if (threadIdx.x < 128) {
        const int r = threadIdx.x >> 3, q = threadIdx.x & 7;
        float acc = b3[q];
        const float* w3 = W3 + (size_t)q * 256;
        #pragma unroll 4
        for (int f = 0; f < DIMD; ++f) acc += m1f[r * DIMD + f] * w3[f];
        #pragma unroll 4
        for (int f = 0; f < DIMD; ++f) acc += m2f[r * DIMD + f] * w3[DIMD + f];
        sc[r * 8 + q] = acc;
    }
    __syncthreads();
    if (threadIdx.x < 16) {
        float mx = sc[threadIdx.x * 8];
        #pragma unroll
        for (int q = 1; q < 8; ++q) mx = fmaxf(mx, sc[threadIdx.x * 8 + q]);
        outRow[rowbase + threadIdx.x] = mx;
    }
}

// --------------------------------------------------------------------------
// Kernel C (1 block): first-occurrence argmax over both score vectors.
// --------------------------------------------------------------------------
__global__ void dec_argmax(const float* __restrict__ s0, const float* __restrict__ s1,
                           int* __restrict__ idx) {
    __shared__ float bv[256];
    __shared__ int   bi[256];
    const int t = threadIdx.x;
    for (int side = 0; side < 2; ++side) {
        const float* s = side ? s1 : s0;
        float best = -3.4e38f; int bidx = 0;
        for (int i = t; i < MLEN; i += 256) {
            float v = s[i];
            if (v > best) { best = v; bidx = i; }
        }
        bv[t] = best; bi[t] = bidx;
        __syncthreads();
        for (int stride = 128; stride > 0; stride >>= 1) {
            if (t < stride) {
                float ov = bv[t + stride]; int oi = bi[t + stride];
                if (ov > bv[t] || (ov == bv[t] && oi < bi[t])) { bv[t] = ov; bi[t] = oi; }
            }
            __syncthreads();
        }
        if (t == 0) idx[side] = bi[0];
        __syncthreads();
    }
}

// --------------------------------------------------------------------------
extern "C" void kernel_launch(void* const* d_in, const int* in_sizes, int n_in,
                              void* d_out, int out_size, void* d_ws, size_t ws_size,
                              hipStream_t stream) {
    (void)in_sizes; (void)n_in; (void)out_size; (void)ws_size;

    const float* U    = (const float*)d_in[0];
    const float* Wih  = (const float*)d_in[1];
    const float* Whh  = (const float*)d_in[2];
    const float* bih  = (const float*)d_in[3];
    const float* bhh  = (const float*)d_in[4];
    const float* WrS  = (const float*)d_in[5];
    const float* brS  = (const float*)d_in[6];
    const float* WgS  = (const float*)d_in[7];
    const float* bgS  = (const float*)d_in[8];
    const float* WeS  = (const float*)d_in[9];
    const float* beS  = (const float*)d_in[10];
    const float* W2S  = (const float*)d_in[11];
    const float* b2S  = (const float*)d_in[12];
    const float* W3S  = (const float*)d_in[13];
    const float* b3S  = (const float*)d_in[14];
    const float* WrE  = (const float*)d_in[15];
    const float* brE  = (const float*)d_in[16];
    const float* WgE  = (const float*)d_in[17];
    const float* bgE  = (const float*)d_in[18];
    const float* WeE  = (const float*)d_in[19];
    const float* beE  = (const float*)d_in[20];
    const float* W2E  = (const float*)d_in[21];
    const float* b2E  = (const float*)d_in[22];
    const float* W3E  = (const float*)d_in[23];
    const float* b3E  = (const float*)d_in[24];

    float* out = (float*)d_out;

    // workspace layout (bytes)
    char* ws = (char*)d_ws;
    size_t off = 0;
    __bf16* Ubf   = (__bf16*)(ws + off); off += (size_t)MLEN * 256 * 2;        // 2 MB
    __bf16* WeSbf = (__bf16*)(ws + off); off += (size_t)2 * 1024 * 384 * 2;    // 1.5 MB
    __bf16* WeEbf = (__bf16*)(ws + off); off += (size_t)2 * 1024 * 384 * 2;
    __bf16* W2Sbf = (__bf16*)(ws + off); off += (size_t)1024 * DIMD * 2;       // 256 KB
    __bf16* W2Ebf = (__bf16*)(ws + off); off += (size_t)1024 * DIMD * 2;
    float*  hbuf  = (float*)(ws + off);  off += 512;
    float*  cbuf  = (float*)(ws + off);  off += 512;
    __bf16* rSbf  = (__bf16*)(ws + off); off += 256;
    __bf16* rEbf  = (__bf16*)(ws + off); off += 256;
    float*  gateS = (float*)(ws + off);  off += 256;
    float*  gateE = (float*)(ws + off);  off += 256;
    int*    idx   = (int*)(ws + off);    off += 256;

    dec_precvt<<<2048, 256, 0, stream>>>(U, WeS, WeE, W2S, W2E,
                                         Ubf, WeSbf, WeEbf, W2Sbf, W2Ebf,
                                         hbuf, cbuf, idx);

    for (int t = 0; t < NSTEPS; ++t) {
        dec_step_small<<<1, 256, 0, stream>>>(U, Wih, Whh, bih, bhh,
                                              WrS, brS, WgS, bgS,
                                              WrE, brE, WgE, bgE,
                                              hbuf, cbuf, rSbf, rEbf,
                                              gateS, gateE, idx);
        float* outS = out + (size_t)t * MLEN;                         // starts (1,4,M)
        float* outE = out + (size_t)NSTEPS * MLEN + (size_t)t * MLEN; // ends   (1,4,M)
        dec_expert_hmn<<<MLEN / 16, 256, 0, stream>>>(Ubf, rSbf, WeSbf, gateS, beS,
                                                      W2Sbf, b2S, W3S, b3S, outS);
        dec_expert_hmn<<<MLEN / 16, 256, 0, stream>>>(Ubf, rEbf, WeEbf, gateE, beE,
                                                      W2Ebf, b2E, W3E, b3E, outE);
        dec_argmax<<<1, 256, 0, stream>>>(outS, outE, idx);
    }
}
